// TransformerDecoder_60851096650131
// MI455X (gfx1250) — matrix operand order repaired and checked by reading, not compiled, not run on hardware
//
#include <hip/hip_runtime.h>
#include <hip/hip_bf16.h>

// ---------------------------------------------------------------- types
typedef __bf16 bf16;
typedef __bf16 bf16x16 __attribute__((ext_vector_type(16)));
typedef __bf16 bf16x8  __attribute__((ext_vector_type(8)));
typedef float  f32x8   __attribute__((ext_vector_type(8)));
typedef __attribute__((address_space(3))) bf16 lds_bf16_t;

#define B_    4
#define S_    2048
#define D_    512
#define H_    8
#define DK_   64
#define DFF_  2048
#define L_    6
#define MTOK  (B_ * S_)          // 8192 token rows

__device__ __forceinline__ bf16 f2bf(float f) {
  unsigned u = __builtin_bit_cast(unsigned, f);
  unsigned r = (u + 0x7FFFu + ((u >> 16) & 1u)) >> 16;   // round-to-nearest-even
  unsigned short s = (unsigned short)r;
  return __builtin_bit_cast(bf16, s);
}

// ---- CDNA5 async global->LDS 16B copy (per lane), tracked by ASYNCcnt
__device__ __forceinline__ void async_copy_b128(const bf16* gsrc, bf16* ldst) {
  unsigned loff = (unsigned)(unsigned long long)(lds_bf16_t*)ldst;
  unsigned long long g = (unsigned long long)(size_t)gsrc;
  asm volatile("global_load_async_to_lds_b128 %0, %1, off"
               :: "v"(loff), "v"(g) : "memory");
}
__device__ __forceinline__ void wait_async0() {
#if __has_builtin(__builtin_amdgcn_s_wait_asynccnt)
  __builtin_amdgcn_s_wait_asynccnt(0);
#else
  asm volatile("s_wait_asynccnt 0x0" ::: "memory");
#endif
}

// ---------------------------------------------------------------- fp32 -> bf16 convert
__global__ __launch_bounds__(256) void convert_bf16_kernel(
    const float* __restrict__ src, bf16* __restrict__ dst, int n) {
  int stride = gridDim.x * blockDim.x;
  for (int i = blockIdx.x * blockDim.x + threadIdx.x; i < n; i += stride)
    dst[i] = f2bf(src[i]);
}

// ---------------------------------------------------------------- fp32 (K,N) -> bf16 (N,K) transpose
// grid (N/32, K/32, L); per-layer matrices of identical shape stacked along z.
__global__ __launch_bounds__(256) void convt_kernel(
    const float* __restrict__ W, bf16* __restrict__ Wt, int K, int N) {
  __shared__ float t[32][33];
  const size_t lay = (size_t)blockIdx.z * K * N;
  W  += lay;  Wt += lay;
  int n0 = blockIdx.x * 32, k0 = blockIdx.y * 32;
  int li = threadIdx.x & 31, wi = threadIdx.x >> 5;
#pragma unroll
  for (int p = 0; p < 4; ++p)
    t[wi + p * 8][li] = W[(size_t)(k0 + wi + p * 8) * N + n0 + li];
  __syncthreads();
#pragma unroll
  for (int p = 0; p < 4; ++p)
    Wt[(size_t)(n0 + wi + p * 8) * K + k0 + li] = f2bf(t[li][wi + p * 8]);
}

// ---------------------------------------------------------------- embedding + positional encoding
__global__ __launch_bounds__(256) void embed_pe_kernel(
    const int* __restrict__ tok, const float* __restrict__ emb,
    float* __restrict__ xF, bf16* __restrict__ xB) {
  int idx = blockIdx.x;                  // token row in [0, B*S)
  int s   = idx % S_;
  int t   = tok[idx];
  size_t base = (size_t)idx * D_;
  const float scale = 22.62741699796952f;   // sqrt(512)
#pragma unroll
  for (int c = 0; c < 2; ++c) {
    int i = threadIdx.x + 256 * c;
    float e  = emb[(size_t)t * D_ + i] * scale;
    float ex = 2.0f * floorf((float)i * 0.5f) * (1.0f / (float)D_);
    float ang = (float)s * __powf(10000.0f, -ex);
    float pe = (i & 1) ? __cosf(ang) : __sinf(ang);
    float v = e + pe;
    xF[base + i] = v;
    xB[base + i] = f2bf(v);
  }
}

// ---------------------------------------------------------------- residual add + LayerNorm
__device__ __forceinline__ float block_sum256(float v, float* red, int tid) {
#pragma unroll
  for (int o = 16; o >= 1; o >>= 1) v += __shfl_xor(v, o, 32);
  if ((tid & 31) == 0) red[tid >> 5] = v;
  __syncthreads();
  float s = 0.f;
#pragma unroll
  for (int i = 0; i < 8; ++i) s += red[i];
  __syncthreads();
  return s;
}

__global__ __launch_bounds__(256) void add_ln_kernel(
    const float* __restrict__ a, const float* __restrict__ res,
    const float* __restrict__ g, const float* __restrict__ bta,
    float* __restrict__ outF, bf16* __restrict__ outB) {
  __shared__ float red[8];
  int row = blockIdx.x, tid = threadIdx.x;
  size_t base = (size_t)row * D_;
  float v0 = a[base + tid]       + res[base + tid];
  float v1 = a[base + tid + 256] + res[base + tid + 256];
  float mean = block_sum256(v0 + v1, red, tid) * (1.0f / (float)D_);
  float d0 = v0 - mean, d1 = v1 - mean;
  float var = block_sum256(d0 * d0 + d1 * d1, red, tid) * (1.0f / (float)D_);
  float inv = rsqrtf(var + 1e-6f);
  float o0 = g[tid]       * d0 * inv + bta[tid];
  float o1 = g[tid + 256] * d1 * inv + bta[tid + 256];
  outF[base + tid] = o0;        outF[base + tid + 256] = o1;
  outB[base + tid] = f2bf(o0);  outB[base + tid + 256] = f2bf(o1);
}

// ---------------------------------------------------------------- WMMA bf16 GEMM
// C[M,N] = A[M,K](bf16) @ Wt[N,K](bf16, pre-transposed) + bias; optional ReLU.
// Outputs: f32 and/or bf16 row-major, optional head-transposed bf16 (B*H,dk,S).
// Block tile 128x128, 8 wave32 waves (4x2) -> 32x64 per wave = 2x4 WMMA accs.
// K stepped by 32 through double-buffered LDS filled by async-to-LDS loads,
// prefetch of step k+1 overlapped with WMMA of step k.
#define BM 128
#define BN 128
#define BK 32

__global__ __launch_bounds__(256) void gemm_bf16_kernel(
    const bf16* __restrict__ A, const bf16* __restrict__ Wt,
    const float* __restrict__ bias,
    float* __restrict__ outF, bf16* __restrict__ outB, bf16* __restrict__ outBT,
    int M, int N, int K, int relu) {
  __shared__ __align__(32) bf16 As[2][BM * BK];   // 2 x 8 KB, [m][k]
  __shared__ __align__(32) bf16 Bs[2][BN * BK];   // 2 x 8 KB, [n][k]

  const int tid  = threadIdx.x;
  const int lane = tid & 31;
  const int wave = tid >> 5;
  const int wm   = wave & 3;
  const int wn   = wave >> 2;
  const int half = lane >> 4;
  const int l16  = lane & 15;
  const int m_blk = blockIdx.y * BM;
  const int n_blk = blockIdx.x * BN;

  // straight-line staging: each thread owns two 16B chunks per tensor
  const int srow = tid >> 2, scc = (tid & 3) * 8;
  const bf16* gA0 = A  + (size_t)(m_blk + srow) * K + scc;
  const bf16* gA1 = gA0 + (size_t)64 * K;
  const bf16* gB0 = Wt + (size_t)(n_blk + srow) * K + scc;
  const bf16* gB1 = gB0 + (size_t)64 * K;
  const int l0 = srow * BK + scc, l1 = (64 + srow) * BK + scc;
  auto stage = [&](int buf, int k0) {
    async_copy_b128(gA0 + k0, &As[buf][l0]);
    async_copy_b128(gA1 + k0, &As[buf][l1]);
    async_copy_b128(gB0 + k0, &Bs[buf][l0]);
    async_copy_b128(gB1 + k0, &Bs[buf][l1]);
  };

  f32x8 acc[2][4];
#pragma unroll
  for (int i = 0; i < 2; ++i)
#pragma unroll
    for (int j = 0; j < 4; ++j) acc[i][j] = (f32x8)0.0f;

  const int nsteps = K / BK;
  stage(0, 0);
  wait_async0();
  __syncthreads();

  for (int st = 0; st < nsteps; ++st) {
    const int cur = st & 1;
    if (st + 1 < nsteps) stage(cur ^ 1, (st + 1) * BK);   // overlap with compute

    bf16x16 af[2];
#pragma unroll
    for (int mt = 0; mt < 2; ++mt) {
      const bf16* ap = &As[cur][(wm * 32 + mt * 16 + l16) * BK];
      bf16x8 lo = *reinterpret_cast<const bf16x8*>(ap + 16 * half);
      bf16x8 hi = *reinterpret_cast<const bf16x8*>(ap + 16 * half + 8);
#pragma unroll
      for (int i = 0; i < 8; ++i) { af[mt][i] = lo[i]; af[mt][8 + i] = hi[i]; }
    }
#pragma unroll
    for (int nt = 0; nt < 4; ++nt) {
      const bf16* bp = &Bs[cur][(wn * 64 + nt * 16 + l16) * BK + 16 * half];
      bf16x16 bfr = *reinterpret_cast<const bf16x16*>(bp);
#pragma unroll
      for (int mt = 0; mt < 2; ++mt)
        acc[mt][nt] = __builtin_amdgcn_wmma_f32_16x16x32_bf16(
            false, af[mt], false, bfr, (short)0, acc[mt][nt], false, false);
    }
    wait_async0();     // prefetch done; compute above hid its latency
    __syncthreads();
  }

  // epilogue: C layout -> row = 8*half + r, col = l16 within each 16x16 tile
#pragma unroll
  for (int mt = 0; mt < 2; ++mt)
#pragma unroll
    for (int nt = 0; nt < 4; ++nt) {
      int col = n_blk + wn * 64 + nt * 16 + l16;
      float bv = bias ? bias[col] : 0.0f;
#pragma unroll
      for (int r = 0; r < 8; ++r) {
        int row = m_blk + wm * 32 + mt * 16 + 8 * half + r;
        float v = acc[mt][nt][r] + bv;
        if (relu) v = fmaxf(v, 0.0f);
        size_t o = (size_t)row * N + col;
        if (outF) outF[o] = v;
        if (outB) outB[o] = f2bf(v);
        if (outBT) {   // (B*H, dk, S) head-transposed copy (N==D_ projections)
          int b = row >> 11, s = row & (S_ - 1);
          int h = col >> 6,  d = col & (DK_ - 1);
          outBT[(((size_t)b * H_ + h) * DK_ + d) * S_ + s] = f2bf(v);
        }
      }
    }
}

// ---------------------------------------------------------------- flash attention (WMMA)
// One block = (b,h, 128 query rows); 8 waves, each owns 16 query rows.
// K staged [kv][dk], V staged from head-transposed global as [dk][kv], both
// double-buffered with async-to-LDS prefetch of the next kv chunk.
#define QT 128
#define KT 32

__global__ __launch_bounds__(256) void flash_attn_kernel(
    const bf16* __restrict__ Q, const bf16* __restrict__ Kp,
    const bf16* __restrict__ VT, bf16* __restrict__ O, int causal) {
  __shared__ __align__(32) bf16 Ks[2][KT * DK_];    // 2 x 4 KB [kv][dk]
  __shared__ __align__(32) bf16 VsT[2][DK_ * KT];   // 2 x 4 KB [dk][kv]
  __shared__ __align__(32) bf16 Ps[8][16 * KT];     // 8 KB per-wave P staging

  const int bh = blockIdx.y, b = bh / H_, h = bh % H_;
  const int qt = blockIdx.x;
  const int tid = threadIdx.x, lane = tid & 31, wave = tid >> 5;
  const int half = lane >> 4, l16 = lane & 15;
  const size_t baseBH = (size_t)b * S_ * D_ + (size_t)h * DK_;
  const size_t baseVT = (size_t)bh * DK_ * S_;
  const int q0 = qt * QT + wave * 16;

  // straight-line staging: one 16B chunk per thread per tensor
  const bf16* gK = Kp + baseBH + (size_t)(tid >> 3) * D_ + (tid & 7) * 8;
  const bf16* gV = VT + baseVT + (size_t)(tid >> 2) * S_ + (tid & 3) * 8;
  const int lK = (tid >> 3) * DK_ + (tid & 7) * 8;
  const int lV = (tid >> 2) * KT + (tid & 3) * 8;
  auto stage = [&](int buf, int kv0) {
    async_copy_b128(gK + (size_t)kv0 * D_, &Ks[buf][lK]);
    async_copy_b128(gV + kv0, &VsT[buf][lV]);
  };

  // q fragments (2 x 16x32 over dk=64), contiguous 16B chunks from global
  bf16x16 qf[2];
  {
    const bf16* qp = Q + baseBH + (size_t)(q0 + l16) * D_;
#pragma unroll
    for (int kc = 0; kc < 2; ++kc) {
      bf16x8 lo = *reinterpret_cast<const bf16x8*>(qp + kc * 32 + 16 * half);
      bf16x8 hi = *reinterpret_cast<const bf16x8*>(qp + kc * 32 + 16 * half + 8);
#pragma unroll
      for (int i = 0; i < 8; ++i) { qf[kc][i] = lo[i]; qf[kc][8 + i] = hi[i]; }
    }
  }

  f32x8 accv[4];
#pragma unroll
  for (int i = 0; i < 4; ++i) accv[i] = (f32x8)0.0f;
  float mrow[8], lrow[8];
#pragma unroll
  for (int r = 0; r < 8; ++r) { mrow[r] = -1e30f; lrow[r] = 0.0f; }

  const int nkv = causal ? (qt + 1) * QT : S_;
  const int nchunks = nkv / KT;
  stage(0, 0);
  wait_async0();
  __syncthreads();

  for (int ch = 0; ch < nchunks; ++ch) {
    const int cur = ch & 1;
    const int kv0 = ch * KT;
    if (ch + 1 < nchunks) stage(cur ^ 1, kv0 + KT);   // overlap with compute

    // scores 16x32: two 16x16 tiles, each accumulating over dk=64 (2 WMMAs)
    f32x8 sc[2];
#pragma unroll
    for (int j = 0; j < 2; ++j) {
      sc[j] = (f32x8)0.0f;
#pragma unroll
      for (int kc = 0; kc < 2; ++kc) {
        bf16x16 bfr = *reinterpret_cast<const bf16x16*>(
            &Ks[cur][(j * 16 + l16) * DK_ + kc * 32 + 16 * half]);
        sc[j] = __builtin_amdgcn_wmma_f32_16x16x32_bf16(
            false, qf[kc], false, bfr, (short)0, sc[j], false, false);
      }
    }

    // scale + causal mask + online softmax (row reductions across 16 lanes)
#pragma unroll
    for (int r = 0; r < 8; ++r) {
      int qrow = q0 + 8 * half + r;
#pragma unroll
      for (int j = 0; j < 2; ++j) {
        float s = sc[j][r] * 0.125f;   // 1/sqrt(64)
        int kcol = kv0 + j * 16 + l16;
        if (causal && kcol > qrow) s = -1e30f;
        sc[j][r] = s;
      }
      float mx = fmaxf(sc[0][r], sc[1][r]);
#pragma unroll
      for (int o = 8; o >= 1; o >>= 1) mx = fmaxf(mx, __shfl_xor(mx, o, 16));
      float mnew = fmaxf(mrow[r], mx);
      float corr = __expf(mrow[r] - mnew);
      float p0 = __expf(sc[0][r] - mnew);
      float p1 = __expf(sc[1][r] - mnew);
      sc[0][r] = p0; sc[1][r] = p1;
      float ps = p0 + p1;
#pragma unroll
      for (int o = 8; o >= 1; o >>= 1) ps += __shfl_xor(ps, o, 16);
      lrow[r] = lrow[r] * corr + ps;
      mrow[r] = mnew;
#pragma unroll
      for (int nt = 0; nt < 4; ++nt) accv[nt][r] *= corr;
    }

    // stage P (C-layout) to LDS, reload in A-fragment layout
#pragma unroll
    for (int j = 0; j < 2; ++j)
#pragma unroll
      for (int r = 0; r < 8; ++r)
        Ps[wave][(8 * half + r) * KT + j * 16 + l16] = f2bf(sc[j][r]);

    bf16x16 pf;
    {
      const bf16* pp = &Ps[wave][l16 * KT];
      bf16x8 lo = *reinterpret_cast<const bf16x8*>(pp + 16 * half);
      bf16x8 hi = *reinterpret_cast<const bf16x8*>(pp + 16 * half + 8);
#pragma unroll
      for (int i = 0; i < 8; ++i) { pf[i] = lo[i]; pf[8 + i] = hi[i]; }
    }

    // acc += P(16x32) @ V(32x64): V fragment contiguous over kv from [dk][kv]
#pragma unroll
    for (int nt = 0; nt < 4; ++nt) {
      bf16x16 vf = *reinterpret_cast<const bf16x16*>(
          &VsT[cur][(nt * 16 + l16) * KT + 16 * half]);
      accv[nt] = __builtin_amdgcn_wmma_f32_16x16x32_bf16(
          false, pf, false, vf, (short)0, accv[nt], false, false);
    }
    wait_async0();
    __syncthreads();
  }

  // write O (bf16), normalized
#pragma unroll
  for (int nt = 0; nt < 4; ++nt)
#pragma unroll
    for (int r = 0; r < 8; ++r) {
      int row = q0 + 8 * half + r;
      float v = accv[nt][r] / lrow[r];
      O[baseBH + (size_t)row * D_ + nt * 16 + l16] = f2bf(v);
    }
}

// ---------------------------------------------------------------- host orchestration
extern "C" void kernel_launch(void* const* d_in, const int* in_sizes, int n_in,
                              void* d_out, int out_size, void* d_ws, size_t ws_size,
                              hipStream_t stream) {
  const int*   tok = (const int*)d_in[0];
  const float* enc = (const float*)d_in[1];
  // d_in[2] comb_mask (causal, computed analytically), d_in[3] padding_mask (zero): unused
  const float* emb = (const float*)d_in[4];
  const float* Wq1 = (const float*)d_in[5];  const float* bq1 = (const float*)d_in[6];
  const float* Wo1 = (const float*)d_in[7];  const float* bo1 = (const float*)d_in[8];
  const float* Wq2 = (const float*)d_in[9];  const float* bq2 = (const float*)d_in[10];
  const float* Wo2 = (const float*)d_in[11]; const float* bo2 = (const float*)d_in[12];
  const float* W1  = (const float*)d_in[13]; const float* b1  = (const float*)d_in[14];
  const float* W2  = (const float*)d_in[15]; const float* b2  = (const float*)d_in[16];
  const float* lng = (const float*)d_in[17]; const float* lnb = (const float*)d_in[18];

  // -------- workspace carve-out
  char* p = (char*)d_ws;
  auto carve = [&](size_t bytes) -> void* {
    void* r = (void*)p;
    p += (bytes + 255) & ~(size_t)255;
    return r;
  };
  const size_t NW_DD  = (size_t)L_ * D_ * D_;
  const size_t NW_DF  = (size_t)L_ * D_ * DFF_;
  const size_t NTOKD  = (size_t)MTOK * D_;
  bf16* Wq1t = (bf16*)carve(NW_DD * 2);      // all weights stored transposed (N,K)
  bf16* Wo1t = (bf16*)carve(NW_DD * 2);
  bf16* Wq2t = (bf16*)carve(NW_DD * 2);
  bf16* Wo2t = (bf16*)carve(NW_DD * 2);
  bf16* W1t  = (bf16*)carve(NW_DF * 2);
  bf16* W2t  = (bf16*)carve(NW_DF * 2);
  bf16* encB = (bf16*)carve(NTOKD * 2);
  float* xF  = (float*)carve(NTOKD * 4);
  float* x1F = (float*)carve(NTOKD * 4);
  float* x2F = (float*)carve(NTOKD * 4);
  float* gF  = (float*)carve(NTOKD * 4);
  bf16* xB   = (bf16*)carve(NTOKD * 2);
  bf16* x1B  = (bf16*)carve(NTOKD * 2);
  bf16* x2B  = (bf16*)carve(NTOKD * 2);
  bf16* Pb   = (bf16*)carve(NTOKD * 2);   // shared q/k/v projection (row-major)
  bf16* PbT  = (bf16*)carve(NTOKD * 2);   // same, head-transposed (B*H, dk, S)
  bf16* KV2b = (bf16*)carve(NTOKD * 2);   // cross k/v projection (row-major)
  bf16* KV2bT= (bf16*)carve(NTOKD * 2);   // head-transposed
  bf16* AOb  = (bf16*)carve(NTOKD * 2);   // attention outputs
  bf16* Hb   = (bf16*)carve((size_t)MTOK * DFF_ * 2);
  (void)ws_size; (void)n_in; (void)in_sizes; (void)out_size;

  // -------- one-time-per-call conversions (weights transposed to (N,K))
  auto convt = [&](const float* s, bf16* d, int K, int N) {
    convt_kernel<<<dim3(N / 32, K / 32, L_), 256, 0, stream>>>(s, d, K, N);
  };
  convt(Wq1, Wq1t, D_, D_);   convt(Wo1, Wo1t, D_, D_);
  convt(Wq2, Wq2t, D_, D_);   convt(Wo2, Wo2t, D_, D_);
  convt(W1,  W1t,  D_, DFF_); convt(W2,  W2t,  DFF_, D_);
  convert_bf16_kernel<<<1024, 256, 0, stream>>>(enc, encB, (int)NTOKD);

  // -------- embedding + positional encoding
  embed_pe_kernel<<<MTOK, 256, 0, stream>>>(tok, emb, xF, xB);

  auto gemm = [&](const bf16* A, const bf16* Wt, const float* bias,
                  float* oF, bf16* oB, bf16* oBT, int M, int N, int K, int relu) {
    gemm_bf16_kernel<<<dim3(N / BN, M / BM), 256, 0, stream>>>(
        A, Wt, bias, oF, oB, oBT, M, N, K, relu);
  };
  const dim3 agrid(S_ / QT, B_ * H_);

  for (int l = 0; l < L_; ++l) {
    const bf16* wq1 = Wq1t + (size_t)l * D_ * D_;
    const bf16* wo1 = Wo1t + (size_t)l * D_ * D_;
    const bf16* wq2 = Wq2t + (size_t)l * D_ * D_;
    const bf16* wo2 = Wo2t + (size_t)l * D_ * D_;
    const bf16* w1  = W1t  + (size_t)l * D_ * DFF_;
    const bf16* w2  = W2t  + (size_t)l * DFF_ * D_;
    const float* lbq1 = bq1 + (size_t)l * D_;
    const float* lbo1 = bo1 + (size_t)l * D_;
    const float* lbq2 = bq2 + (size_t)l * D_;
    const float* lbo2 = bo2 + (size_t)l * D_;
    const float* lb1  = b1  + (size_t)l * DFF_;
    const float* lb2  = b2  + (size_t)l * D_;
    const float* g0 = lng + ((size_t)l * 3 + 0) * D_;
    const float* g1 = lng + ((size_t)l * 3 + 1) * D_;
    const float* g2 = lng + ((size_t)l * 3 + 2) * D_;
    const float* c0 = lnb + ((size_t)l * 3 + 0) * D_;
    const float* c1 = lnb + ((size_t)l * 3 + 1) * D_;
    const float* c2 = lnb + ((size_t)l * 3 + 2) * D_;

    // self-attention (q = k = v: reference reuses Wq for all three)
    gemm(xB, wq1, lbq1, nullptr, Pb, PbT, MTOK, D_, D_, 0);
    flash_attn_kernel<<<agrid, 256, 0, stream>>>(Pb, Pb, PbT, AOb, 1);
    gemm(AOb, wo1, lbo1, gF, nullptr, nullptr, MTOK, D_, D_, 0);
    add_ln_kernel<<<MTOK, 256, 0, stream>>>(gF, xF, g0, c0, x1F, x1B);

    // cross-attention (k = v share one projection of enc)
    gemm(x1B, wq2, lbq2, nullptr, Pb, nullptr, MTOK, D_, D_, 0);
    gemm(encB, wq2, lbq2, nullptr, KV2b, KV2bT, MTOK, D_, D_, 0);
    flash_attn_kernel<<<agrid, 256, 0, stream>>>(Pb, KV2b, KV2bT, AOb, 0);
    gemm(AOb, wo2, lbo2, gF, nullptr, nullptr, MTOK, D_, D_, 0);
    add_ln_kernel<<<MTOK, 256, 0, stream>>>(gF, x1F, g1, c1, x2F, x2B);

    // feed-forward
    gemm(x2B, w1, lb1, nullptr, Hb, nullptr, MTOK, DFF_, D_, 1);
    gemm(Hb, w2, lb2, gF, nullptr, nullptr, MTOK, D_, DFF_, 0);
    float* outF = (l == L_ - 1) ? (float*)d_out : xF;
    add_ln_kernel<<<MTOK, 256, 0, stream>>>(gF, x2F, g2, c2, outF, xB);
  }
}
